// get_model_87875030876906
// MI455X (gfx1250) — compile-verified, matrix operand order checked by reading, not run on hardware
//
#include <hip/hip_runtime.h>
#include <hip/hip_bf16.h>
#include <math.h>

// ---------------------------------------------------------------------------
// Types for CDNA5 WMMA
// ---------------------------------------------------------------------------
typedef __attribute__((ext_vector_type(16))) _Float16 v16h;
typedef __attribute__((ext_vector_type(8)))  float    v8f;

#define ACT_NONE 0
#define ACT_RELU 1
#define ACT_TANH 2

// ---------------------------------------------------------------------------
// FAST GEMM:  C[M,N] = act(A[M,K] @ W[K,N] + bias[N])
// Requires M%16==0, N%16==0, K%32==0.
// Grid: x = ceil(M/16/8) (8 waves per block, one 16-row tile each),
//       y = N/16 (all waves in a block share one 16-col tile of W).
// Per K-step: B tile (32x16) is staged cooperatively into LDS as f16 and
// read back via ds_load; A fragment is four global_load_b128 per lane.
// EXEC is all-ones at every WMMA (tile clamp instead of masking).
// ---------------------------------------------------------------------------
template <int ACT>
__global__ __launch_bounds__(256)
void k_gemm_wmma_fast(const float* __restrict__ A, const float* __restrict__ W,
                      const float* __restrict__ bias, float* __restrict__ C,
                      int M, int N, int K)
{
    const int lane = threadIdx.x & 31;
    const int wave = threadIdx.x >> 5;
    const int tilesM = M >> 4;
    long tmt = (long)blockIdx.x * 8 + wave;
    if (tmt >= tilesM) tmt = tilesM - 1;      // clamp: redundant recompute, no masking
    const int tm = (int)tmt << 4;
    const int tn = blockIdx.y << 4;

    const bool hi   = lane >= 16;
    const int  l15  = lane & 15;
    const int  koff = hi ? 8 : 0;
    const int  arow = tm + l15;
    const int  bcol = tn + l15;

    __shared__ _Float16 bs[32 * 16];          // one shared B tile per block

    const float* arowp = A + (size_t)arow * K;
    const int r0 = threadIdx.x >> 4;          // 0..15
    const int c0 = threadIdx.x & 15;

    v8f acc = {};
    for (int k0 = 0; k0 < K; k0 += 32) {
        __syncthreads();
        // cooperative stage of W[k0:k0+32, tn:tn+16] -> LDS (f16)
        bs[threadIdx.x]       = (_Float16)W[(size_t)(k0 + r0)      * N + tn + c0];
        bs[threadIdx.x + 256] = (_Float16)W[(size_t)(k0 + r0 + 16) * N + tn + c0];
        __syncthreads();

        if (k0 + 32 < K)
            __builtin_prefetch(arowp + k0 + 32, 0, 1);   // global_prefetch_b8

        const float4 a0 = *(const float4*)(arowp + k0 + koff);
        const float4 a1 = *(const float4*)(arowp + k0 + koff + 4);
        const float4 a2 = *(const float4*)(arowp + k0 + 16 + koff);
        const float4 a3 = *(const float4*)(arowp + k0 + 16 + koff + 4);

        v16h a, b;
        a[0] = (_Float16)a0.x; a[1] = (_Float16)a0.y; a[2] = (_Float16)a0.z; a[3] = (_Float16)a0.w;
        a[4] = (_Float16)a1.x; a[5] = (_Float16)a1.y; a[6] = (_Float16)a1.z; a[7] = (_Float16)a1.w;
        a[8] = (_Float16)a2.x; a[9] = (_Float16)a2.y; a[10] = (_Float16)a2.z; a[11] = (_Float16)a2.w;
        a[12] = (_Float16)a3.x; a[13] = (_Float16)a3.y; a[14] = (_Float16)a3.z; a[15] = (_Float16)a3.w;
#pragma unroll
        for (int i = 0; i < 16; ++i) {
            const int kr = koff + (i & 7) + ((i >> 3) << 4);
            b[i] = bs[kr * 16 + l15];
        }
        acc = __builtin_amdgcn_wmma_f32_16x16x32_f16(
                  false, a, false, b, (short)0, acc, false, false);
    }

    const int rbase = tm + (hi ? 8 : 0);
    const float bb = (bias != nullptr) ? bias[bcol] : 0.0f;
#pragma unroll
    for (int i = 0; i < 8; ++i) {
        float v = acc[i] + bb;
        if (ACT == ACT_RELU) v = fmaxf(v, 0.0f);
        if (ACT == ACT_TANH) v = tanhf(v);
        C[(size_t)(rbase + i) * N + bcol] = v;
    }
}

// ---------------------------------------------------------------------------
// GENERIC GEMM (any M,N,K): clamped addresses + selects instead of branchy
// guarded loads, so loads are straight-line and EXEC stays full for WMMA.
// ---------------------------------------------------------------------------
template <int ACT>
__global__ __launch_bounds__(256)
void k_gemm_wmma_gen(const float* __restrict__ A, const float* __restrict__ W,
                     const float* __restrict__ bias, float* __restrict__ C,
                     int M, int N, int K)
{
    const int lane  = threadIdx.x & 31;
    const int wave  = threadIdx.x >> 5;
    const int tilesN = (N + 15) >> 4;
    const long tilesM = (long)((M + 15) >> 4);
    const long tile = (long)blockIdx.x * 8 + wave;
    if (tile >= tilesM * tilesN) return;          // wave-uniform branch
    const int tm = (int)(tile / tilesN) * 16;
    const int tn = (int)(tile % tilesN) * 16;

    const bool hi   = lane >= 16;
    const int  l15  = lane & 15;
    const int  koff = hi ? 8 : 0;
    const int  arow = tm + l15;
    const int  bcol = tn + l15;
    const int  arc  = arow < M ? arow : M - 1;    // clamped (always in-bounds)
    const int  bcc  = bcol < N ? bcol : N - 1;

    v8f acc = {};
    for (int k0 = 0; k0 < K; k0 += 32) {
        v16h a, b;
#pragma unroll
        for (int i = 0; i < 16; ++i) {
            const int kk = k0 + (i & 7) + koff + ((i >> 3) << 4);
            const int kc = kk < K ? kk : K - 1;
            float av = A[(size_t)arc * K + kc];
            float bv = W[(size_t)kc * N + bcc];
            av = (arow < M && kk < K) ? av : 0.0f;
            bv = (bcol < N && kk < K) ? bv : 0.0f;
            a[i] = (_Float16)av;
            b[i] = (_Float16)bv;
        }
        acc = __builtin_amdgcn_wmma_f32_16x16x32_f16(
                  false, a, false, b, (short)0, acc, false, false);
    }

    const int rbase = tm + (hi ? 8 : 0);
    const float bb = (bias != nullptr && bcol < N) ? bias[bcol] : 0.0f;
#pragma unroll
    for (int i = 0; i < 8; ++i) {
        const int r = rbase + i;
        if (r < M && bcol < N) {
            float v = acc[i] + bb;
            if (ACT == ACT_RELU) v = fmaxf(v, 0.0f);
            if (ACT == ACT_TANH) v = tanhf(v);
            C[(size_t)r * N + bcol] = v;
        }
    }
}

// ---------------------------------------------------------------------------
// Furthest point sampling: one block per batch, dist[] lives in LDS.
// ---------------------------------------------------------------------------
__global__ __launch_bounds__(256)
void k_fps(const float* __restrict__ xyz, int* __restrict__ fidx,
           int N, int npoint)
{
    const int b = blockIdx.x;
    const int t = threadIdx.x;
    __shared__ float dist[2048];
    __shared__ float rv[256];
    __shared__ int   ri[256];
    __shared__ int   s_far;

    for (int i = t; i < N; i += 256) dist[i] = 1e10f;
    if (t == 0) s_far = 0;
    __syncthreads();

    for (int it = 0; it < npoint; ++it) {
        const int far = s_far;
        if (t == 0) fidx[(size_t)b * npoint + it] = far;
        const float cx = xyz[((size_t)b * N + far) * 3 + 0];
        const float cy = xyz[((size_t)b * N + far) * 3 + 1];
        const float cz = xyz[((size_t)b * N + far) * 3 + 2];
        float bv = -1.0f; int bi = 0;
        for (int i = t; i < N; i += 256) {
            const float dx = xyz[((size_t)b * N + i) * 3 + 0] - cx;
            const float dy = xyz[((size_t)b * N + i) * 3 + 1] - cy;
            const float dz = xyz[((size_t)b * N + i) * 3 + 2] - cz;
            const float d  = dx * dx + dy * dy + dz * dz;
            const float nd = fminf(dist[i], d);
            dist[i] = nd;
            if (nd > bv) { bv = nd; bi = i; }
        }
        rv[t] = bv; ri[t] = bi;
        __syncthreads();
        for (int k = 128; k > 0; k >>= 1) {
            if (t < k && rv[t + k] > rv[t]) { rv[t] = rv[t + k]; ri[t] = ri[t + k]; }
            __syncthreads();
        }
        if (t == 0) s_far = ri[0];
        __syncthreads();
    }
}

// dst[b, r, :] = src[b, idx[b, r], :]   (idx flattened (B, RS))
__global__ void k_gather_rows(const float* __restrict__ src, const int* __restrict__ idx,
                              float* __restrict__ dst, int N, long RS, int C, long total)
{
    const long id = (long)blockIdx.x * blockDim.x + threadIdx.x;
    if (id >= total) return;
    const long b = id / RS;
    const int  n = idx[id];
    const float* s = src + ((size_t)b * N + n) * C;
    float* d = dst + (size_t)id * C;
    for (int c = 0; c < C; ++c) d[c] = s[c];
}

__global__ void k_ballquery(const float* __restrict__ xyz, const float* __restrict__ nxyz,
                            int* __restrict__ gidx, int N, int M, int ns, float r2, long total)
{
    const long id = (long)blockIdx.x * blockDim.x + threadIdx.x;
    if (id >= total) return;
    const long b = id / M;
    const float qx = nxyz[id * 3 + 0], qy = nxyz[id * 3 + 1], qz = nxyz[id * 3 + 2];
    int* out = gidx + (size_t)id * ns;
    int cnt = 0, first = -1;
    for (int n = 0; n < N && cnt < ns; ++n) {
        const float dx = xyz[((size_t)b * N + n) * 3 + 0] - qx;
        const float dy = xyz[((size_t)b * N + n) * 3 + 1] - qy;
        const float dz = xyz[((size_t)b * N + n) * 3 + 2] - qz;
        if (dx * dx + dy * dy + dz * dz <= r2) {
            if (first < 0) first = n;
            out[cnt++] = n;
        }
    }
    if (first < 0) first = 0;
    for (; cnt < ns; ++cnt) out[cnt] = first;
}

// g[b,m,s, 0:3] = xyz[gidx] - new_xyz ; g[b,m,s, 3:3+C] = feats[gidx]
__global__ void k_group(const float* __restrict__ xyz, const float* __restrict__ nxyz,
                        const float* __restrict__ feats, const int* __restrict__ gidx,
                        float* __restrict__ g, int N, int M, int S, int C, long total)
{
    const long id = (long)blockIdx.x * blockDim.x + threadIdx.x;
    if (id >= total) return;
    const long bm = id / S;
    const long b  = bm / M;
    const int  n  = gidx[id];
    const float* p = xyz  + ((size_t)b * N + n) * 3;
    const float* q = nxyz + (size_t)bm * 3;
    float* o = g + (size_t)id * (3 + C);
    o[0] = p[0] - q[0]; o[1] = p[1] - q[1]; o[2] = p[2] - q[2];
    const float* f = feats + ((size_t)b * N + n) * C;
    for (int c = 0; c < C; ++c) o[3 + c] = f[c];
}

// per-channel mean/var over R rows (jnp.var, ddof=0)
__global__ __launch_bounds__(256)
void k_colstats(const float* __restrict__ X, float* __restrict__ mean,
                float* __restrict__ var, long R, int C)
{
    const int c = blockIdx.x, t = threadIdx.x;
    float s = 0.0f, s2 = 0.0f;
    for (long r = t; r < R; r += 256) {
        const float v = X[(size_t)r * C + c];
        s += v; s2 += v * v;
    }
    __shared__ float sh[256], sh2[256];
    sh[t] = s; sh2[t] = s2;
    __syncthreads();
    for (int k = 128; k > 0; k >>= 1) {
        if (t < k) { sh[t] += sh[t + k]; sh2[t] += sh2[t + k]; }
        __syncthreads();
    }
    if (t == 0) {
        const float m = sh[0] / (float)R;
        mean[c] = m;
        var[c]  = sh2[0] / (float)R - m * m;
    }
}

__global__ void k_bn_relu(float* __restrict__ X, const float* __restrict__ mean,
                          const float* __restrict__ var, int C, long total)
{
    const long id = (long)blockIdx.x * blockDim.x + threadIdx.x;
    if (id >= total) return;
    const int c = (int)(id % C);
    float v = (X[id] - mean[c]) * rsqrtf(var[c] + 1e-5f);
    X[id] = v > 0.0f ? v : 0.0f;
}

// Y[r,c] = max_s X[r,s,c]
__global__ void k_max_mid(const float* __restrict__ X, float* __restrict__ Y,
                          int S, int C, long total)
{
    const long id = (long)blockIdx.x * blockDim.x + threadIdx.x;
    if (id >= total) return;
    const long r = id / C;
    const int  c = (int)(id % C);
    const float* p = X + (size_t)r * S * C + c;
    float m = -1e30f;
    for (int s = 0; s < S; ++s) m = fmaxf(m, p[(size_t)s * C]);
    Y[id] = m;
}

// k-nearest-neighbors: per-query insertion sort held in LDS (blockDim=64)
__global__ __launch_bounds__(64)
void k_knn(const float* __restrict__ xyz, int* __restrict__ knn,
           int N, int K, long total)
{
    __shared__ float sd[64 * 64];
    __shared__ int   si[64 * 64];
    const long id = (long)blockIdx.x * 64 + threadIdx.x;
    if (id >= total) return;
    const long b = id / N;
    float* bd = sd + threadIdx.x * 64;
    int*   bi = si + threadIdx.x * 64;
    for (int i = 0; i < K; ++i) { bd[i] = 1e30f; bi[i] = 0; }
    const float qx = xyz[id * 3 + 0], qy = xyz[id * 3 + 1], qz = xyz[id * 3 + 2];
    for (int j = 0; j < N; ++j) {
        const float dx = xyz[((size_t)b * N + j) * 3 + 0] - qx;
        const float dy = xyz[((size_t)b * N + j) * 3 + 1] - qy;
        const float dz = xyz[((size_t)b * N + j) * 3 + 2] - qz;
        const float d  = dx * dx + dy * dy + dz * dz;
        if (d < bd[K - 1]) {
            int pos = K - 1;
            while (pos > 0 && bd[pos - 1] > d) {
                bd[pos] = bd[pos - 1]; bi[pos] = bi[pos - 1]; --pos;
            }
            bd[pos] = d; bi[pos] = j;
        }
    }
    int* o = knn + (size_t)id * K;
    for (int i = 0; i < K; ++i) o[i] = bi[i];
}

// out[b,n,k,:] = xyz[b,n,:] - xyz[b, knn[b,n,k], :]
__global__ void k_knn_diff3(const float* __restrict__ xyz, const int* __restrict__ knn,
                            float* __restrict__ out, int N, int K, long total)
{
    const long id = (long)blockIdx.x * blockDim.x + threadIdx.x;
    if (id >= total) return;
    const long bn = id / K;
    const long b  = bn / N;
    const int  j  = knn[id];
    const float* p = xyz + (size_t)bn * 3;
    const float* q = xyz + ((size_t)b * N + j) * 3;
    float* o = out + (size_t)id * 3;
    o[0] = p[0] - q[0]; o[1] = p[1] - q[1]; o[2] = p[2] - q[2];
}

// out[bn,s,c] = q[bn,c] - kk[bn,s,c] + pos[bn,s,c]
__global__ void k_attn_in(const float* __restrict__ q, const float* __restrict__ kk,
                          const float* __restrict__ pos, float* __restrict__ out,
                          int S, int C, long total)
{
    const long id = (long)blockIdx.x * blockDim.x + threadIdx.x;
    if (id >= total) return;
    const int  c   = (int)(id % C);
    const long bn  = (id / C) / S;
    out[id] = q[(size_t)bn * C + c] - kk[id] + pos[id];
}

// softmax over middle axis S of X:(R,S,C), with pre-scale 1/sqrt(dk)
__global__ void k_softmax_mid(float* __restrict__ X, int S, int C, float inv, long total)
{
    const long id = (long)blockIdx.x * blockDim.x + threadIdx.x;
    if (id >= total) return;
    const long r = id / C;
    const int  c = (int)(id % C);
    float* p = X + (size_t)r * S * C + c;
    float m = -1e30f;
    for (int s = 0; s < S; ++s) m = fmaxf(m, p[(size_t)s * C] * inv);
    float sum = 0.0f;
    for (int s = 0; s < S; ++s) sum += expf(p[(size_t)s * C] * inv - m);
    const float rs = 1.0f / sum;
    for (int s = 0; s < S; ++s) {
        const size_t o = (size_t)s * C;
        p[o] = expf(p[o] * inv - m) * rs;
    }
}

// out[r,c] = sum_s A[r,s,c] * W[r,s,c]
__global__ void k_wsum_mid(const float* __restrict__ A, const float* __restrict__ Wt,
                           float* __restrict__ out, int S, int C, long total)
{
    const long id = (long)blockIdx.x * blockDim.x + threadIdx.x;
    if (id >= total) return;
    const long r = id / C;
    const int  c = (int)(id % C);
    const float* pa = A  + (size_t)r * S * C + c;
    const float* pw = Wt + (size_t)r * S * C + c;
    float acc = 0.0f;
    for (int s = 0; s < S; ++s) acc += pa[(size_t)s * C] * pw[(size_t)s * C];
    out[id] = acc;
}

__global__ void k_add(const float* __restrict__ a, const float* __restrict__ b,
                      float* __restrict__ o, long total)
{
    const long id = (long)blockIdx.x * blockDim.x + threadIdx.x;
    if (id < total) o[id] = a[id] + b[id];
}

// global (x - mean)/std with ddof=1, single block
__global__ __launch_bounds__(256)
void k_gnorm(float* __restrict__ x, int n)
{
    __shared__ float s1[256], s2[256];
    const int t = threadIdx.x;
    float a = 0.0f, b = 0.0f;
    for (int i = t; i < n; i += 256) { const float v = x[i]; a += v; b += v * v; }
    s1[t] = a; s2[t] = b;
    __syncthreads();
    for (int k = 128; k > 0; k >>= 1) {
        if (t < k) { s1[t] += s1[t + k]; s2[t] += s2[t + k]; }
        __syncthreads();
    }
    const float m   = s1[0] / (float)n;
    const float var = (s2[0] - (float)n * m * m) / (float)(n - 1);
    const float inv = rsqrtf(var);
    for (int i = t; i < n; i += 256) x[i] = (x[i] - m) * inv;
}

__global__ void k_add_noise(const float* __restrict__ x, const float* __restrict__ noise,
                            const int* __restrict__ snr, float* __restrict__ y, long total)
{
    const long id = (long)blockIdx.x * blockDim.x + threadIdx.x;
    if (id >= total) return;
    const float f = sqrtf(powf(10.0f, -(float)(*snr) / 10.0f));
    y[id] = x[id] + noise[id] * f;
}

// nxyz[(bn*ratio+r)*3+d] = xyz[bn*3+d] + off[(bn*ratio+r)*3+d] * radius
__global__ void k_up_xyz(const float* __restrict__ xyz, const float* __restrict__ off,
                         float* __restrict__ nxyz, int ratio, float radius, long total)
{
    const long id = (long)blockIdx.x * blockDim.x + threadIdx.x;
    if (id >= total) return;
    const int  d  = (int)(id % 3);
    const long bn = (id / 3) / ratio;
    nxyz[id] = xyz[bn * 3 + d] + off[id] * radius;
}

// rowmin[b,i] = min_j ||A[b,i] - Bp[b,j]||^2
__global__ void k_nnmin(const float* __restrict__ A, const float* __restrict__ Bp,
                        float* __restrict__ rowmin, int NA, int NB, long total)
{
    const long id = (long)blockIdx.x * blockDim.x + threadIdx.x;
    if (id >= total) return;
    const long b = id / NA;
    const float ax = A[id * 3 + 0], ay = A[id * 3 + 1], az = A[id * 3 + 2];
    const float* base = Bp + (size_t)b * NB * 3;
    float m = 1e30f;
    for (int j = 0; j < NB; ++j) {
        const float dx = base[j * 3 + 0] - ax;
        const float dy = base[j * 3 + 1] - ay;
        const float dz = base[j * 3 + 2] - az;
        m = fminf(m, dx * dx + dy * dy + dz * dz);
    }
    rowmin[id] = m;
}

__global__ __launch_bounds__(256)
void k_cd_reduce(const float* __restrict__ r1, const float* __restrict__ r2,
                 int n, float scale, float* __restrict__ out)
{
    __shared__ float sh[256];
    const int t = threadIdx.x;
    float s = 0.0f;
    for (int i = t; i < n; i += 256) s += r1[i] + r2[i];
    sh[t] = s;
    __syncthreads();
    for (int k = 128; k > 0; k >>= 1) {
        if (t < k) sh[t] += sh[t + k];
        __syncthreads();
    }
    if (t == 0) out[0] = sh[0] * scale;
}

// ---------------------------------------------------------------------------
// Host-side orchestration
// ---------------------------------------------------------------------------
namespace {

struct Bump {
    char* base; size_t off;
    float* f(size_t n) { off = (off + 255) & ~(size_t)255; float* p = (float*)(base + off); off += n * 4; return p; }
    int*   i(size_t n) { off = (off + 255) & ~(size_t)255; int*   p = (int*)(base + off);   off += n * 4; return p; }
};

inline unsigned gsz(long total, int block) { return (unsigned)((total + block - 1) / block); }

inline void gemm(hipStream_t st, const float* A, const float* W, const float* b,
                 float* C, long M, int N, int K, int act)
{
    if ((M & 15) == 0 && (N & 15) == 0 && (K & 31) == 0) {
        const dim3 grid((unsigned)((M / 16 + 7) / 8), (unsigned)(N / 16));
        if (act == ACT_RELU)
            k_gemm_wmma_fast<ACT_RELU><<<grid, 256, 0, st>>>(A, W, b, C, (int)M, N, K);
        else if (act == ACT_TANH)
            k_gemm_wmma_fast<ACT_TANH><<<grid, 256, 0, st>>>(A, W, b, C, (int)M, N, K);
        else
            k_gemm_wmma_fast<ACT_NONE><<<grid, 256, 0, st>>>(A, W, b, C, (int)M, N, K);
    } else {
        const long tiles = ((M + 15) / 16) * (long)((N + 15) / 16);
        const dim3 grid((unsigned)((tiles + 7) / 8));
        if (act == ACT_RELU)
            k_gemm_wmma_gen<ACT_RELU><<<grid, 256, 0, st>>>(A, W, b, C, (int)M, N, K);
        else if (act == ACT_TANH)
            k_gemm_wmma_gen<ACT_TANH><<<grid, 256, 0, st>>>(A, W, b, C, (int)M, N, K);
        else
            k_gemm_wmma_gen<ACT_NONE><<<grid, 256, 0, st>>>(A, W, b, C, (int)M, N, K);
    }
}

// Parameter leaf order: jax pytree flatten (dicts sorted by key, 'W' < 'b').
// attn block (15): d1W d1b d2W d2b fc1W fc1b fc2W fc2b g1W g1b g2W g2b wkW wqW wvW
#define P_ATTN1 0
#define P_ATTN2 15
#define P_ATTN3 30
#define P_COOR  45
#define P_DEC   51
#define P_SA1   57
#define P_SA2   63
#define P_UP1   69   // fW fb o1W o1b o2W o2b
#define P_UP2   75
#define P_XYZ   81
#define P_NOISE 82
#define P_SNR   83

float* tblock(hipStream_t st, void* const* d_in, int P, const float* xyz,
              const float* feats, int Bn, int Nn, int Kk, int dm, Bump& ws)
{
    const float* d1W  = (const float*)d_in[P + 0];
    const float* d1b  = (const float*)d_in[P + 1];
    const float* d2W  = (const float*)d_in[P + 2];
    const float* d2b  = (const float*)d_in[P + 3];
    const float* fc1W = (const float*)d_in[P + 4];
    const float* fc1b = (const float*)d_in[P + 5];
    const float* fc2W = (const float*)d_in[P + 6];
    const float* fc2b = (const float*)d_in[P + 7];
    const float* g1W  = (const float*)d_in[P + 8];
    const float* g1b  = (const float*)d_in[P + 9];
    const float* g2W  = (const float*)d_in[P + 10];
    const float* g2b  = (const float*)d_in[P + 11];
    const float* wkW  = (const float*)d_in[P + 12];
    const float* wqW  = (const float*)d_in[P + 13];
    const float* wvW  = (const float*)d_in[P + 14];

    const long R  = (long)Bn * Nn;
    const long RS = R * Kk;

    int* knn = ws.i(RS);
    k_knn<<<gsz(R, 64), 64, 0, st>>>(xyz, knn, Nn, Kk, R);

    float* x = ws.f(R * dm);
    gemm(st, feats, fc1W, fc1b, x, R, dm, dm, ACT_NONE);
    float* q  = ws.f(R * dm); gemm(st, x, wqW, nullptr, q,  R, dm, dm, ACT_NONE);
    float* xk = ws.f(R * dm); gemm(st, x, wkW, nullptr, xk, R, dm, dm, ACT_NONE);
    float* xv = ws.f(R * dm); gemm(st, x, wvW, nullptr, xv, R, dm, dm, ACT_NONE);

    float* kk = ws.f(RS * dm);
    k_gather_rows<<<gsz(RS, 256), 256, 0, st>>>(xk, knn, kk, Nn, (long)Nn * Kk, dm, RS);
    float* v = ws.f(RS * dm);
    k_gather_rows<<<gsz(RS, 256), 256, 0, st>>>(xv, knn, v, Nn, (long)Nn * Kk, dm, RS);

    float* diff = ws.f(RS * 3);
    k_knn_diff3<<<gsz(RS, 256), 256, 0, st>>>(xyz, knn, diff, Nn, Kk, RS);
    float* p1  = ws.f(RS * dm); gemm(st, diff, d1W, d1b, p1, RS, dm, 3, ACT_RELU);
    float* pos = ws.f(RS * dm); gemm(st, p1, d2W, d2b, pos, RS, dm, dm, ACT_NONE);

    float* ain = ws.f(RS * dm);
    k_attn_in<<<gsz(RS * dm, 256), 256, 0, st>>>(q, kk, pos, ain, Kk, dm, RS * dm);
    float* a1   = ws.f(RS * dm); gemm(st, ain, g1W, g1b, a1, RS, dm, dm, ACT_RELU);
    float* attn = ws.f(RS * dm); gemm(st, a1, g2W, g2b, attn, RS, dm, dm, ACT_NONE);
    k_softmax_mid<<<gsz(R * dm, 256), 256, 0, st>>>(attn, Kk, dm,
                                                    1.0f / sqrtf((float)dm), R * dm);

    float* vp = ws.f(RS * dm);
    k_add<<<gsz(RS * dm, 256), 256, 0, st>>>(v, pos, vp, RS * dm);
    float* res = ws.f(R * dm);
    k_wsum_mid<<<gsz(R * dm, 256), 256, 0, st>>>(attn, vp, res, Kk, dm, R * dm);

    float* o = ws.f(R * dm);
    gemm(st, res, fc2W, fc2b, o, R, dm, dm, ACT_NONE);
    float* out = ws.f(R * dm);
    k_add<<<gsz(R * dm, 256), 256, 0, st>>>(o, feats, out, R * dm);
    return out;
}

void set_abs(hipStream_t st, void* const* d_in, int P, const float* xyz,
             const float* feats, int Bn, int Nn, int Cin, int npoint,
             float radius, int ns, const int* dims, Bump& ws,
             float** out_xyz, float** out_f)
{
    const long BM = (long)Bn * npoint;
    int* fidx = ws.i(BM);
    k_fps<<<Bn, 256, 0, st>>>(xyz, fidx, Nn, npoint);

    float* nxyz = ws.f(BM * 3);
    k_gather_rows<<<gsz(BM, 256), 256, 0, st>>>(xyz, fidx, nxyz, Nn, npoint, 3, BM);

    int* gidx = ws.i(BM * ns);
    k_ballquery<<<gsz(BM, 256), 256, 0, st>>>(xyz, nxyz, gidx, Nn, npoint, ns,
                                              radius * radius, BM);

    const long RS = BM * ns;
    const int  C0 = 3 + Cin;
    float* g = ws.f(RS * C0);
    k_group<<<gsz(RS, 256), 256, 0, st>>>(xyz, nxyz, feats, gidx, g, Nn, npoint, ns, Cin, RS);

    float* mean = ws.f(512);
    float* var  = ws.f(512);
    float* cur = g; int curC = C0;
    for (int l = 0; l < 3; ++l) {
        const float* W  = (const float*)d_in[P + 2 * l + 0];
        const float* bb = (const float*)d_in[P + 2 * l + 1];
        const int Co = dims[l];
        float* nxt = ws.f(RS * Co);
        gemm(st, cur, W, bb, nxt, RS, Co, curC, ACT_NONE);
        k_colstats<<<Co, 256, 0, st>>>(nxt, mean, var, RS, Co);
        k_bn_relu<<<gsz(RS * Co, 256), 256, 0, st>>>(nxt, mean, var, Co, RS * Co);
        cur = nxt; curC = Co;
    }
    float* fout = ws.f(BM * curC);
    k_max_mid<<<gsz(BM * curC, 256), 256, 0, st>>>(cur, fout, ns, curC, BM * curC);
    *out_xyz = nxyz;
    *out_f   = fout;
}

} // namespace

extern "C" void kernel_launch(void* const* d_in, const int* in_sizes, int n_in,
                              void* d_out, int out_size, void* d_ws, size_t ws_size,
                              hipStream_t stream)
{
    (void)in_sizes; (void)n_in; (void)out_size; (void)ws_size;
    Bump ws{(char*)d_ws, 0};

    const float* xyz   = (const float*)d_in[P_XYZ];
    const float* noise = (const float*)d_in[P_NOISE];
    const int*   snr   = (const int*)d_in[P_SNR];
    float* out_x2 = (float*)d_out;                       // 16*2048*3
    float* out_cd = out_x2 + (size_t)16 * 2048 * 3;      // scalar

    // --- SA1: 2048 -> 512 points, MLP 6->64->64->128 ---
    float *l1_xyz, *l1_f;
    const int sa1dims[3] = {64, 64, 128};
    set_abs(stream, d_in, P_SA1, xyz, xyz, 16, 2048, 3, 512, 0.2f, 32,
            sa1dims, ws, &l1_xyz, &l1_f);

    // --- transformer block 1 (N=512, k=32, dm=128) ---
    float* l1f2 = tblock(stream, d_in, P_ATTN1, l1_xyz, l1_f, 16, 512, 32, 128, ws);

    // --- SA2: 512 -> 128 points, MLP 131->128->128->256 ---
    float *l2_xyz, *l2_f;
    const int sa2dims[3] = {128, 128, 256};
    set_abs(stream, d_in, P_SA2, l1_xyz, l1f2, 16, 512, 128, 128, 0.4f, 64,
            sa2dims, ws, &l2_xyz, &l2_f);

    // --- transformer block 2 (N=128, k=64, dm=256) ---
    float* x2f = tblock(stream, d_in, P_ATTN2, l2_xyz, l2_f, 16, 128, 64, 256, ws);

    // --- global max over points, global normalize, add noise ---
    float* xg = ws.f(16 * 256);
    k_max_mid<<<gsz(16 * 256, 256), 256, 0, stream>>>(x2f, xg, 128, 256, 16 * 256);
    k_gnorm<<<1, 256, 0, stream>>>(xg, 16 * 256);
    float* y = ws.f(16 * 256);
    k_add_noise<<<gsz(16 * 256, 256), 256, 0, stream>>>(xg, noise, snr, y, 16 * 256);

    // --- decoder: 256 -> 512 -> 512 -> 128*256 ---
    float* h1  = ws.f(16 * 512);
    gemm(stream, y,  (const float*)d_in[P_DEC + 0], (const float*)d_in[P_DEC + 1],
         h1, 16, 512, 256, ACT_RELU);
    float* h2  = ws.f(16 * 512);
    gemm(stream, h1, (const float*)d_in[P_DEC + 2], (const float*)d_in[P_DEC + 3],
         h2, 16, 512, 512, ACT_RELU);
    float* dlf = ws.f((size_t)16 * 128 * 256);
    gemm(stream, h2, (const float*)d_in[P_DEC + 4], (const float*)d_in[P_DEC + 5],
         dlf, 16, 128 * 256, 512, ACT_NONE);

    // --- coordinate MLP: (2048,256) -> 128 -> 128 -> 3 ---
    float* c1 = ws.f((size_t)2048 * 128);
    gemm(stream, dlf, (const float*)d_in[P_COOR + 0], (const float*)d_in[P_COOR + 1],
         c1, 2048, 128, 256, ACT_RELU);
    float* c2 = ws.f((size_t)2048 * 128);
    gemm(stream, c1, (const float*)d_in[P_COOR + 2], (const float*)d_in[P_COOR + 3],
         c2, 2048, 128, 128, ACT_RELU);
    float* nx0 = ws.f((size_t)2048 * 3);
    gemm(stream, c2, (const float*)d_in[P_COOR + 4], (const float*)d_in[P_COOR + 5],
         nx0, 2048, 3, 128, ACT_NONE);

    // --- transformer block 3 (N=128, k=64, dm=256) ---
    float* nf0 = tblock(stream, d_in, P_ATTN3, nx0, dlf, 16, 128, 64, 256, ws);

    // --- upsample 1: 128 -> 512 points ---
    float* u1 = ws.f((size_t)2048 * 256);
    gemm(stream, nf0, (const float*)d_in[P_UP1 + 2], (const float*)d_in[P_UP1 + 3],
         u1, 2048, 256, 256, ACT_RELU);
    float* off1 = ws.f((size_t)2048 * 12);
    gemm(stream, u1, (const float*)d_in[P_UP1 + 4], (const float*)d_in[P_UP1 + 5],
         off1, 2048, 12, 256, ACT_TANH);
    float* x1 = ws.f((size_t)16 * 512 * 3);
    k_up_xyz<<<gsz((long)2048 * 12, 256), 256, 0, stream>>>(nx0, off1, x1, 4, 0.1f,
                                                            (long)2048 * 12);
    float* f1 = ws.f((size_t)2048 * 1024);   // == (16*512, 256) after reshape
    gemm(stream, nf0, (const float*)d_in[P_UP1 + 0], (const float*)d_in[P_UP1 + 1],
         f1, 2048, 1024, 256, ACT_RELU);

    // --- upsample 2: 512 -> 2048 points (features of up2 are unused) ---
    float* u2 = ws.f((size_t)8192 * 256);
    gemm(stream, f1, (const float*)d_in[P_UP2 + 2], (const float*)d_in[P_UP2 + 3],
         u2, 8192, 256, 256, ACT_RELU);
    float* off2 = ws.f((size_t)8192 * 12);
    gemm(stream, u2, (const float*)d_in[P_UP2 + 4], (const float*)d_in[P_UP2 + 5],
         off2, 8192, 12, 256, ACT_TANH);
    k_up_xyz<<<gsz((long)8192 * 12, 256), 256, 0, stream>>>(x1, off2, out_x2, 4, 0.1f,
                                                            (long)8192 * 12);

    // --- chamfer distance (deterministic two-stage reduction) ---
    float* r1 = ws.f((size_t)16 * 2048);
    float* r2 = ws.f((size_t)16 * 2048);
    k_nnmin<<<gsz((long)16 * 2048, 256), 256, 0, stream>>>(xyz, out_x2, r1, 2048, 2048,
                                                           (long)16 * 2048);
    k_nnmin<<<gsz((long)16 * 2048, 256), 256, 0, stream>>>(out_x2, xyz, r2, 2048, 2048,
                                                           (long)16 * 2048);
    k_cd_reduce<<<1, 256, 0, stream>>>(r1, r2, 16 * 2048,
                                       1.0f / (float)(16 * 2048), out_cd);
}
